// LinearKernelCrossAttention_23081154248840
// MI455X (gfx1250) — matrix operand order repaired and checked by reading, not compiled, not run on hardware
//
#include <hip/hip_runtime.h>
#include <hip/hip_bf16.h>
#include <math.h>

typedef __bf16 bf16_t;
typedef __attribute__((ext_vector_type(16))) __bf16        v16bf;
typedef __attribute__((ext_vector_type(8)))  float         v8f;
typedef __attribute__((ext_vector_type(4)))  unsigned int  uint4v;
typedef __attribute__((ext_vector_type(4)))  float         float4v;

union FragU { uint4v q[2]; v16bf v; };
union Pack8 { uint4v q; bf16_t e[8]; };
union Pack2 { unsigned int u; bf16_t e[2]; };

#define WMMA_BF16(a, b, c) \
  __builtin_amdgcn_wmma_f32_16x16x32_bf16(false, (a), false, (b), (short)0, (c), false, false)

// ---------------------------------------------------------------------------
// Fragment loaders (CDNA5 wave32 WMMA 16x16x32 bf16 layouts)
//  A (16x32): lane L -> row M = m0+(L&15); K chunks [(L>>4)*8 .. +7] and
//             [16+(L>>4)*8 .. +7]  (two contiguous 16B spans)
//  B (32x16): lane L -> col N = n0+(L&15); same K chunks as A (two 16B spans
//             when B is stored transposed Bt[n][k])
//  C/D: lane L col N = L&15; VGPR r -> row M = r + 8*(L>>4)
// ---------------------------------------------------------------------------
static __device__ inline v16bf frag_a_bf16(const bf16_t* base, int ld, int m0, int k0) {
  const int lane = threadIdx.x & 31;
  const bf16_t* p = base + (size_t)(m0 + (lane & 15)) * ld + k0 + ((lane >> 4) * 8);
  FragU u;
  u.q[0] = *(const uint4v*)(p);
  u.q[1] = *(const uint4v*)(p + 16);
  return u.v;
}

static __device__ inline v16bf frag_a_f32(const float* base, int ld, int m0, int k0) {
  const int lane = threadIdx.x & 31;
  const float* p = base + (size_t)(m0 + (lane & 15)) * ld + k0 + ((lane >> 4) * 8);
  float4v f0 = *(const float4v*)(p);
  float4v f1 = *(const float4v*)(p + 4);
  float4v f2 = *(const float4v*)(p + 16);
  float4v f3 = *(const float4v*)(p + 20);
  v16bf a;
#pragma unroll
  for (int i = 0; i < 4; ++i) {
    a[i]      = (__bf16)f0[i];
    a[4 + i]  = (__bf16)f1[i];
    a[8 + i]  = (__bf16)f2[i];
    a[12 + i] = (__bf16)f3[i];
  }
  return a;
}

// B from transposed storage Bt[n][k] (two 16B spans per lane)
static __device__ inline v16bf frag_bT_bf16(const bf16_t* bt, int ld, int n0, int k0) {
  const int lane = threadIdx.x & 31;
  const bf16_t* p = bt + (size_t)(n0 + (lane & 15)) * ld + k0 + ((lane >> 4) * 8);
  FragU u;
  u.q[0] = *(const uint4v*)(p);
  u.q[1] = *(const uint4v*)(p + 16);
  return u.v;
}

// ---------------------------------------------------------------------------
// Kernel 0: W [in,out] fp32 -> Wt [out,in] bf16
// ---------------------------------------------------------------------------
__global__ void wt_bf16_kernel(const float* __restrict__ W, bf16_t* __restrict__ Wt,
                               int K, int N) {
  int idx = blockIdx.x * blockDim.x + threadIdx.x;
  if (idx < K * N) {
    int k = idx / N, n = idx - k * N;
    Wt[(size_t)n * K + k] = (__bf16)W[idx];
  }
}

// ---------------------------------------------------------------------------
// Kernels 1 & 4: Y = act(X @ W + b)  — M x 512 x 512 GEMM
// block = 256 thr (8 waves); macro tile 256(M) x 64(N); wave -> 32 rows.
// Wt slab (64 N-rows x 512 K, 64KB) staged in LDS via gfx1250 async DMA
// (global_load_async_to_lds_b128 + s_wait_asynccnt), rows padded to 520
// elements to spread DS banks. Epilogue staged through LDS (reusing the
// slab region) for coalesced b128 stores.
// ---------------------------------------------------------------------------
#define WT_LDS_LD 520                         // 512 + 8 pad (row = 1040 B)
template <bool A_BF16, bool OUT_F32, bool ACT_ELU1>
__global__ __launch_bounds__(256) void proj_kernel(const void* __restrict__ Ain,
                                                   const bf16_t* __restrict__ Wt,
                                                   const float* __restrict__ bias,
                                                   void* __restrict__ Yout) {
  __shared__ __align__(16) unsigned char smem[64 * WT_LDS_LD * 2];  // 66560 B
  const int lane = threadIdx.x & 31;
  const int wave = threadIdx.x >> 5;
  const int mBlock = blockIdx.x * 256;
  const int m0 = mBlock + wave * 32;          // wave covers rows m0..m0+31
  const int n0 = blockIdx.y * 64;

  // ---- async-DMA the contiguous 64KB Wt slab into LDS (padded rows) ----
  {
    const bf16_t* slab = Wt + (size_t)n0 * 512;        // 64 contiguous rows
    const unsigned lds_base = (unsigned)(size_t)(void*)smem;
#pragma unroll
    for (int i = 0; i < 16; ++i) {
      const unsigned c = threadIdx.x + 256 * i;        // 4096 chunks of 16B
      const unsigned row = c >> 6;
      const unsigned off = (c & 63) * 8;               // element offset in row
      const unsigned long long ga =
          (unsigned long long)(const void*)(slab + (size_t)row * 512 + off);
      const unsigned la = lds_base + row * (WT_LDS_LD * 2) + off * 2;
      asm volatile("global_load_async_to_lds_b128 %0, %1, off"
                   :: "v"(la), "v"(ga) : "memory");
    }
    asm volatile("s_wait_asynccnt 0" ::: "memory");
    __syncthreads();
  }
  const bf16_t* WtL = (const bf16_t*)smem;

  v8f acc[2][4] = {};
  for (int k0 = 0; k0 < 512; k0 += 32) {
    if (k0 + 32 < 512) {  // prefetch next A k-slab (global_prefetch_b8)
      if constexpr (A_BF16)
        __builtin_prefetch((const bf16_t*)Ain + (size_t)(m0 + (lane & 15)) * 512 + k0 + 32, 0, 1);
      else
        __builtin_prefetch((const float*)Ain + (size_t)(m0 + (lane & 15)) * 512 + k0 + 32, 0, 1);
    }
    v16bf a0, a1;
    if constexpr (A_BF16) {
      a0 = frag_a_bf16((const bf16_t*)Ain, 512, m0, k0);
      a1 = frag_a_bf16((const bf16_t*)Ain, 512, m0 + 16, k0);
    } else {
      a0 = frag_a_f32((const float*)Ain, 512, m0, k0);
      a1 = frag_a_f32((const float*)Ain, 512, m0 + 16, k0);
    }
#pragma unroll
    for (int t = 0; t < 4; ++t) {
      v16bf b = frag_bT_bf16(WtL, WT_LDS_LD, 16 * t, k0);
      acc[0][t] = WMMA_BF16(a0, b, acc[0][t]);
      acc[1][t] = WMMA_BF16(a1, b, acc[1][t]);
    }
  }

  __syncthreads();  // done reading Wt slab; reuse LDS for output staging

  // bias + activation, stage 256x64 tile into LDS
#pragma unroll
  for (int s = 0; s < 2; ++s) {
#pragma unroll
    for (int t = 0; t < 4; ++t) {
      const int nl = 16 * t + (lane & 15);
      const float bv = bias[n0 + nl];
#pragma unroll
      for (int r = 0; r < 8; ++r) {
        const int ml = wave * 32 + s * 16 + r + 8 * (lane >> 4);
        float v = acc[s][t][r] + bv;
        if constexpr (ACT_ELU1) v = (v > 0.f) ? (v + 1.f) : __expf(v);  // elu(x)+1
        if constexpr (OUT_F32) ((float*)smem)[ml * 64 + nl] = v;
        else                   ((bf16_t*)smem)[ml * 64 + nl] = (__bf16)v;
      }
    }
  }
  __syncthreads();

  // coalesced cooperative store
  if constexpr (OUT_F32) {
    const float* st = (const float*)smem;
    float* outp = (float*)Yout;
#pragma unroll
    for (int i = 0; i < 16; ++i) {
      const int c = threadIdx.x + 256 * i;  // 4096 chunks of 4 floats
      const int row = c >> 4;
      const int off = (c & 15) * 4;
      *(float4v*)(outp + (size_t)(mBlock + row) * 512 + n0 + off) =
          *(const float4v*)(st + row * 64 + off);
    }
  } else {
    const bf16_t* st = (const bf16_t*)smem;
    bf16_t* outp = (bf16_t*)Yout;
#pragma unroll
    for (int i = 0; i < 8; ++i) {
      const int c = threadIdx.x + 256 * i;  // 2048 chunks of 8 bf16
      const int row = c >> 3;
      const int off = (c & 7) * 8;
      *(uint4v*)(outp + (size_t)(mBlock + row) * 512 + n0 + off) =
          *(const uint4v*)(st + row * 64 + off);
    }
  }
}

// ---------------------------------------------------------------------------
// Kernel 2: per (b,s,h): KtV[64,64] = K^T V over a=2048 ; ksum[64] = sum_a K
// K/V tiles staged transposed in LDS ([d][a+pad]) -> all frags contiguous.
// KtV stored TRANSPOSED ([dv][dk]) so kernel 3's B-frags are contiguous.
// ---------------------------------------------------------------------------
__global__ __launch_bounds__(256) void ktv_kernel(const bf16_t* __restrict__ kact,
                                                  const bf16_t* __restrict__ vact,
                                                  bf16_t* __restrict__ ktv,
                                                  float* __restrict__ ksum) {
  __shared__ __align__(16) bf16_t Kt[64 * 40];  // [d][a], row stride 40 (pad 8)
  __shared__ __align__(16) bf16_t Vt[64 * 40];
  const int bsh = blockIdx.x;                   // [0, 512)
  const int bs = bsh >> 3, h = bsh & 7;
  const bf16_t* Kp = kact + (size_t)bs * 2048 * 512 + h * 64;
  const bf16_t* Vp = vact + (size_t)bs * 2048 * 512 + h * 64;
  const int lane = threadIdx.x & 31;
  const int wave = threadIdx.x >> 5;
  const int mt = wave >> 1;                     // 0..3
  const int nt0 = (wave & 1) * 2;               // 0 or 2

  const int sa = threadIdx.x >> 3;              // staging: a row 0..31
  const int sd = (threadIdx.x & 7) * 8;         // staging: d col 0..56

  v8f acc0 = {}, acc1 = {};
  for (int a0 = 0; a0 < 2048; a0 += 32) {
    {  // coalesced b128 loads, transpose-scatter into LDS
      Pack8 kk, vv;
      kk.q = *(const uint4v*)(Kp + (size_t)(a0 + sa) * 512 + sd);
      vv.q = *(const uint4v*)(Vp + (size_t)(a0 + sa) * 512 + sd);
#pragma unroll
      for (int j = 0; j < 8; ++j) {
        Kt[(sd + j) * 40 + sa] = kk.e[j];
        Vt[(sd + j) * 40 + sa] = vv.e[j];
      }
    }
    __syncthreads();
    v16bf a  = frag_a_bf16(Kt, 40, mt * 16, 0);        // A = K^T
    v16bf b0 = frag_bT_bf16(Vt, 40, (nt0 + 0) * 16, 0);
    v16bf b1 = frag_bT_bf16(Vt, 40, (nt0 + 1) * 16, 0);
    acc0 = WMMA_BF16(a, b0, acc0);
    acc1 = WMMA_BF16(a, b1, acc1);
    __syncthreads();
  }

  // store KtV transposed: ktvT[n(dv)][m(dk)]
  bf16_t* Dp = ktv + (size_t)bsh * 64 * 64;
#pragma unroll
  for (int r = 0; r < 8; ++r) {
    const int m = mt * 16 + r + 8 * (lane >> 4);
    Dp[(size_t)((nt0 + 0) * 16 + (lane & 15)) * 64 + m] = (__bf16)acc0[r];
    Dp[(size_t)((nt0 + 1) * 16 + (lane & 15)) * 64 + m] = (__bf16)acc1[r];
  }

  // ksum: 8 a-slices x 32 column-pairs, coalesced 32-bit loads, LDS reduce
  __syncthreads();
  {
    float* red = (float*)Kt;                    // reuse LDS (needs 2KB)
    const int d2 = (threadIdx.x & 31) * 2;
    const int part = threadIdx.x >> 5;          // 0..7
    float s0 = 0.f, s1 = 0.f;
    for (int a = part * 256; a < part * 256 + 256; ++a) {
      Pack2 w;
      w.u = *(const unsigned int*)(Kp + (size_t)a * 512 + d2);
      s0 += (float)w.e[0];
      s1 += (float)w.e[1];
    }
    red[part * 64 + d2]     = s0;
    red[part * 64 + d2 + 1] = s1;
    __syncthreads();
    if (threadIdx.x < 64) {
      float t = 0.f;
#pragma unroll
      for (int p = 0; p < 8; ++p) t += red[p * 64 + threadIdx.x];
      ksum[(size_t)bsh * 64 + threadIdx.x] = t;
    }
  }
}

// ---------------------------------------------------------------------------
// Kernel 3: per (b,s,h) M-tile: V = Z * (q @ KtV), Z = 1/(q . ksum + eps)
// ---------------------------------------------------------------------------
__global__ __launch_bounds__(256) void qktv_kernel(const bf16_t* __restrict__ qact,
                                                   const bf16_t* __restrict__ ktvT,
                                                   const float* __restrict__ ksum,
                                                   bf16_t* __restrict__ vout) {
  __shared__ float KsS[64];
  __shared__ float Zs[128];
  __shared__ __align__(16) bf16_t stage[128 * 64];
  const int bsh = blockIdx.x;
  const int bs = bsh >> 3, h = bsh & 7;
  const bf16_t* Qp = qact + (size_t)bs * 2048 * 512 + h * 64;
  const bf16_t* Bp = ktvT + (size_t)bsh * 4096;    // [dv][dk]
  bf16_t* Op = vout + (size_t)bs * 2048 * 512 + h * 64;
  const int lane = threadIdx.x & 31;
  const int wave = threadIdx.x >> 5;
  const int mBlock = blockIdx.y * 128;

  if (threadIdx.x < 64) KsS[threadIdx.x] = ksum[(size_t)bsh * 64 + threadIdx.x];
  __syncthreads();
  if (threadIdx.x < 128) {
    const bf16_t* qr = Qp + (size_t)(mBlock + threadIdx.x) * 512;
    float d = 0.f;
#pragma unroll
    for (int c = 0; c < 8; ++c) {
      Pack8 pk;
      pk.q = *(const uint4v*)(qr + c * 8);
#pragma unroll
      for (int j = 0; j < 8; ++j) d += (float)pk.e[j] * KsS[c * 8 + j];
    }
    Zs[threadIdx.x] = 1.0f / (d + 1e-6f);
  }
  __syncthreads();

  const int m0 = mBlock + wave * 16;
  v8f acc[4] = {};
#pragma unroll
  for (int k0 = 0; k0 < 64; k0 += 32) {
    v16bf a = frag_a_bf16(Qp, 512, m0, k0);
#pragma unroll
    for (int t = 0; t < 4; ++t) {
      v16bf b = frag_bT_bf16(Bp, 64, 16 * t, k0);
      acc[t] = WMMA_BF16(a, b, acc[t]);
    }
  }
#pragma unroll
  for (int t = 0; t < 4; ++t) {
    const int nl = 16 * t + (lane & 15);
#pragma unroll
    for (int r = 0; r < 8; ++r) {
      const int ml = wave * 16 + r + 8 * (lane >> 4);
      stage[ml * 64 + nl] = (__bf16)(acc[t][r] * Zs[ml]);
    }
  }
  __syncthreads();
#pragma unroll
  for (int i = 0; i < 4; ++i) {
    const int c = threadIdx.x + 256 * i;
    const int row = c >> 3;
    const int off = (c & 7) * 8;
    *(uint4v*)(Op + (size_t)(mBlock + row) * 512 + off) =
        *(const uint4v*)(stage + row * 64 + off);
  }
}

// ---------------------------------------------------------------------------
// Host launcher
// ---------------------------------------------------------------------------
extern "C" void kernel_launch(void* const* d_in, const int* in_sizes, int n_in,
                              void* d_out, int out_size, void* d_ws, size_t ws_size,
                              hipStream_t stream) {
  const float* input_q  = (const float*)d_in[0];
  const float* input_kv = (const float*)d_in[1];
  const float* Wq = (const float*)d_in[2];
  const float* bq = (const float*)d_in[3];
  const float* Wk = (const float*)d_in[4];
  const float* bk = (const float*)d_in[5];
  const float* Wv = (const float*)d_in[6];
  const float* bv = (const float*)d_in[7];
  const float* Wo = (const float*)d_in[8];
  const float* bo = (const float*)d_in[9];

  const int M  = in_sizes[0] / 512;   // tokens: b*s*a = 131072
  const int BS = M / 2048;            // b*s = 64

  char* ws = (char*)d_ws;
  bf16_t* WqT = (bf16_t*)ws;  ws += 512 * 512 * sizeof(bf16_t);
  bf16_t* WkT = (bf16_t*)ws;  ws += 512 * 512 * sizeof(bf16_t);
  bf16_t* WvT = (bf16_t*)ws;  ws += 512 * 512 * sizeof(bf16_t);
  bf16_t* WoT = (bf16_t*)ws;  ws += 512 * 512 * sizeof(bf16_t);
  bf16_t* qact = (bf16_t*)ws; ws += (size_t)M * 512 * sizeof(bf16_t);
  bf16_t* kact = (bf16_t*)ws; ws += (size_t)M * 512 * sizeof(bf16_t);
  bf16_t* vbuf = (bf16_t*)ws; ws += (size_t)M * 512 * sizeof(bf16_t);  // reused for scaled attn out
  bf16_t* ktvT = (bf16_t*)ws; ws += (size_t)BS * 8 * 64 * 64 * sizeof(bf16_t);
  float*  ksum = (float*)ws;  ws += (size_t)BS * 8 * 64 * sizeof(float);

  {
    dim3 g((512 * 512 + 255) / 256), b(256);
    wt_bf16_kernel<<<g, b, 0, stream>>>(Wq, WqT, 512, 512);
    wt_bf16_kernel<<<g, b, 0, stream>>>(Wk, WkT, 512, 512);
    wt_bf16_kernel<<<g, b, 0, stream>>>(Wv, WvT, 512, 512);
    wt_bf16_kernel<<<g, b, 0, stream>>>(Wo, WoT, 512, 512);
  }
  {
    dim3 g(M / 256, 8), b(256);
    proj_kernel<false, false, true ><<<g, b, 0, stream>>>(input_q,  WqT, bq, qact);
    proj_kernel<false, false, true ><<<g, b, 0, stream>>>(input_kv, WkT, bk, kact);
    proj_kernel<false, false, false><<<g, b, 0, stream>>>(input_kv, WvT, bv, vbuf);
  }
  ktv_kernel<<<dim3(BS * 8), dim3(256), 0, stream>>>(kact, vbuf, ktvT, ksum);
  qktv_kernel<<<dim3(BS * 8, 2048 / 128), dim3(256), 0, stream>>>(qact, ktvT, ksum, vbuf);
  proj_kernel<true, true, false><<<dim3(M / 256, 8), dim3(256), 0, stream>>>(vbuf, WoT, bo, (float*)d_out);
}
